// HeteroEncoderHAN_78881369358658
// MI455X (gfx1250) — matrix operand order repaired and checked by reading, not compiled, not run on hardware
//
#include <hip/hip_runtime.h>
#include <cstddef>

// ---------------- problem constants (from reference) ----------------
#define NI_N   100000
#define ND_N   50000
#define HID_C  128
#define E_CO_N 500000
#define E_UI_N 300000
#define E_PW_N 300000
#define K_MASK 64
#define LAT_N  64

typedef __bf16 bf16_t;
typedef bf16_t v16bf __attribute__((ext_vector_type(16)));
typedef bf16_t v8bf  __attribute__((ext_vector_type(8)));
typedef float  v8f   __attribute__((ext_vector_type(8)));

// pack two fp32 into one dword of bf16 pairs (lowers to v_cvt_pk_bf16_f32)
__device__ __forceinline__ unsigned pack_bf16(float lo, float hi) {
    union { bf16_t h[2]; unsigned u; } r;
    r.h[0] = (bf16_t)lo; r.h[1] = (bf16_t)hi;
    return r.u;
}

// =====================================================================
// Tiled GEMM: C[M,N] = act(A[M,K] @ B[K,N] + bias)
// A,B fp32 row-major. REQUIRES N % 64 == 0 and K % 32 == 0
// (true for every call site: N in {128,64}, K in {128,768}).
// Block 256 thr (8 waves); tile 64x64; K step 32; per-wave 16x32 strip.
// LDS: A bf16 [64][40]  -> frag = 2x ds_load_b128 + reg concat;
//      B K-pair packed, TRANSPOSED [n][pk] dwords [64][20]
//         -> frag = 2x ds_load_b128 straight into operand regs
//            (dword v of lane half h = {bf16(K=16h+2v), bf16(K=16h+2v+1)}: the
//             same contiguous k = 16h+s order as A; no shuffles, no movs).
// ACT: 0 none, 1 relu, 2 score mode (no C store; accumulates
//      sum_n sum_h tanh(c+bias)*q into *score_out).
// =====================================================================
template<int ACT>
__launch_bounds__(256)
__global__ void gemm_wmma_bf16(const float* __restrict__ A, const float* __restrict__ B,
                               const float* __restrict__ bias, float* __restrict__ C,
                               int M, int N, int Kd, const int* __restrict__ rowidx,
                               const float* __restrict__ qv, float* __restrict__ score_out)
{
    __shared__ bf16_t   As[64][40];    // 64 rows x 32 K (+pad), 80B row stride
    __shared__ unsigned BsT[64][20];   // [n][packed-K 0..15] (+pad), 80B row stride

    const int tid  = threadIdx.x;
    const int lane = tid & 31;
    const int wave = tid >> 5;
    const int wm   = wave & 3;
    const int wn   = wave >> 2;
    const int m0   = blockIdx.y * 64;
    const int n0   = blockIdx.x * 64;

    v8f acc0 = {}; v8f acc1 = {};

    // A staging: 8 consecutive K per thread
    const int ar = tid >> 2;                 // 0..63
    const int ac = (tid & 3) << 3;           // 0,8,16,24
    // B staging: one packed-K row (2 K) x 4 N per thread
    const int pr = tid >> 4;                 // 0..15
    const int pc = (tid & 15) << 2;          // 0..60

    const bool   arok = (m0 + ar) < M;
    const size_t arow = arok ? (size_t)(rowidx ? rowidx[m0 + ar] : (m0 + ar)) : 0;

    // fragment addresses
    const int am  = (wm << 4) + (lane & 15);
    const int kh  = (lane >> 4) << 4;        // 0 or 16
    const int bn  = (wn << 5) + (lane & 15);
    const int pr0 = (lane >> 4) << 3;        // packed-K base: 0 or 8

    for (int k0 = 0; k0 < Kd; k0 += 32) {
        // ---- stage A (fp32 -> packed bf16, one b128 store) ----
        {
            uint4 w;
            if (arok) {
                const float4* p = (const float4*)(A + arow * (size_t)Kd + (size_t)(k0 + ac));
                float4 v0 = p[0], v1 = p[1];
                w.x = pack_bf16(v0.x, v0.y); w.y = pack_bf16(v0.z, v0.w);
                w.z = pack_bf16(v1.x, v1.y); w.w = pack_bf16(v1.z, v1.w);
            } else {
                w.x = w.y = w.z = w.w = 0u;
            }
            *(uint4*)&As[ar][ac] = w;
        }
        // ---- stage B (two K rows -> packed dwords, transposed scatter) ----
        {
            const float4 r0 = *(const float4*)(B + (size_t)(k0 + 2*pr    ) * N + (size_t)(n0 + pc));
            const float4 r1 = *(const float4*)(B + (size_t)(k0 + 2*pr + 1) * N + (size_t)(n0 + pc));
            BsT[pc + 0][pr] = pack_bf16(r0.x, r1.x);
            BsT[pc + 1][pr] = pack_bf16(r0.y, r1.y);
            BsT[pc + 2][pr] = pack_bf16(r0.z, r1.z);
            BsT[pc + 3][pr] = pack_bf16(r0.w, r1.w);
        }
        __syncthreads();

        // ---- A fragment: two contiguous 8-element runs (2x ds_load_b128) ----
        v8bf alo = *(const v8bf*)&As[am][kh];
        v8bf ahi = *(const v8bf*)&As[am][kh + 8];
        v16bf af = __builtin_shufflevector(alo, ahi,
                    0,1,2,3,4,5,6,7,8,9,10,11,12,13,14,15);

        // ---- B fragments: contiguous packed dwords (2x ds_load_b128 each) ----
        union { v16bf v; uint4 q[2]; } fb0, fb1;
        fb0.q[0] = *(const uint4*)&BsT[bn][pr0];
        fb0.q[1] = *(const uint4*)&BsT[bn][pr0 + 4];
        fb1.q[0] = *(const uint4*)&BsT[bn + 16][pr0];
        fb1.q[1] = *(const uint4*)&BsT[bn + 16][pr0 + 4];

        acc0 = __builtin_amdgcn_wmma_f32_16x16x32_bf16(false, af, false, fb0.v, (short)0, acc0, false, false);
        acc1 = __builtin_amdgcn_wmma_f32_16x16x32_bf16(false, af, false, fb1.v, (short)0, acc1, false, false);
        __syncthreads();
    }

    // ---- epilogue (C layout: VGPR r -> row (lane>>4)*8 + r, col lane&15) ----
    const int cn = n0 + (wn << 5) + (lane & 15);
    const int rb = m0 + (wm << 4) + ((lane >> 4) << 3);

    if (ACT == 2) {
        float partial = 0.0f;
        const float b0 = bias ? bias[cn]      : 0.0f;
        const float b1 = bias ? bias[cn + 16] : 0.0f;
        const float q0 = qv[cn], q1 = qv[cn + 16];
        #pragma unroll
        for (int r = 0; r < 8; ++r) {
            if (rb + r < M) {
                partial += tanhf(acc0[r] + b0) * q0;
                partial += tanhf(acc1[r] + b1) * q1;
            }
        }
        #pragma unroll
        for (int off = 16; off > 0; off >>= 1)
            partial += __shfl_xor(partial, off, 32);
        if (lane == 0) atomicAdd(score_out, partial);
    } else {
        const float b0 = bias ? bias[cn]      : 0.0f;
        const float b1 = bias ? bias[cn + 16] : 0.0f;
        #pragma unroll
        for (int r = 0; r < 8; ++r) {
            const int row = rb + r;
            if (row < M) {
                float v0 = acc0[r] + b0;
                float v1 = acc1[r] + b1;
                if (ACT == 1) { v0 = fmaxf(v0, 0.0f); v1 = fmaxf(v1, 0.0f); }
                C[(size_t)row * N + cn]      = v0;
                C[(size_t)row * N + cn + 16] = v1;
            }
        }
    }
}

// ---------------- per-head attention dots: out[n,h] = <h[n,h*32:],att[h,:]> ----
__global__ void head_score(const float* __restrict__ h, const float* __restrict__ att,
                           float* __restrict__ out, int Nn)
{
    int idx = blockIdx.x * blockDim.x + threadIdx.x;   // n*4 + head
    if (idx >= Nn * 4) return;
    int n = idx >> 2, hd = idx & 3;
    const float* hp = h + (size_t)n * HID_C + hd * 32;
    const float* ap = att + hd * 32;
    float s = 0.0f;
    #pragma unroll
    for (int d = 0; d < 32; d += 4) {
        float4 a = *(const float4*)(hp + d);
        float4 b = *(const float4*)(ap + d);
        s += a.x*b.x + a.y*b.y + a.z*b.z + a.w*b.w;
    }
    out[idx] = s;
}

// ---------------- order-preserving float<->uint for atomic max ----------------
__device__ __forceinline__ unsigned encf(float f) {
    unsigned u = __float_as_uint(f);
    return (u & 0x80000000u) ? ~u : (u | 0x80000000u);
}
__device__ __forceinline__ float decf(unsigned u) {
    return (u & 0x80000000u) ? __uint_as_float(u & 0x7fffffffu) : __uint_as_float(~u);
}
#define ENC_NEG_INF 0x007FFFFFu

// edge pass A: alpha = leaky_relu(a_src[s]+a_dst[d], 0.2); store; atomic segment-max
__global__ void edge_alpha_max(const int* __restrict__ ei, int E,
                               const float* __restrict__ asrc, const float* __restrict__ adst,
                               float* __restrict__ ebuf, unsigned* __restrict__ amax)
{
    int e = blockIdx.x * blockDim.x + threadIdx.x;
    if (e >= E) return;
    int s = ei[e], d = ei[E + e];
    #pragma unroll
    for (int h = 0; h < 4; ++h) {
        float al = asrc[s*4 + h] + adst[d*4 + h];
        al = al > 0.0f ? al : 0.2f * al;
        ebuf[(size_t)e*4 + h] = al;
        atomicMax(&amax[d*4 + h], encf(al));
    }
}

// edge pass B: e = exp(alpha - amax[dst]); atomic segment-sum
__global__ void edge_exp_sum(const int* __restrict__ ei, int E,
                             float* __restrict__ ebuf, const unsigned* __restrict__ amax,
                             float* __restrict__ ssum)
{
    int e = blockIdx.x * blockDim.x + threadIdx.x;
    if (e >= E) return;
    int d = ei[E + e];
    #pragma unroll
    for (int h = 0; h < 4; ++h) {
        float m = decf(amax[d*4 + h]);
        if (m < -1.0e37f) m = 0.0f;                 // isfinite guard (only -inf possible)
        float ex = __expf(ebuf[(size_t)e*4 + h] - m);
        ebuf[(size_t)e*4 + h] = ex;
        atomicAdd(&ssum[d*4 + h], ex);
    }
}

// edge pass C: wave-per-edge coalesced message scatter (128 floats per edge)
__global__ void edge_scatter(const int* __restrict__ ei, int E,
                             const float* __restrict__ hsrc, const float* __restrict__ ebuf,
                             const float* __restrict__ ssum, float* __restrict__ agg)
{
    int w = (int)((blockIdx.x * (unsigned)blockDim.x + threadIdx.x) >> 5);
    int lane = threadIdx.x & 31;
    if (w >= E) return;
    int s = ei[w], d = ei[E + w];
    #pragma unroll
    for (int i = 0; i < 4; ++i) {            // comp c = lane + 32*i -> head i
        int c = lane + 32*i;
        float wgt = ebuf[(size_t)w*4 + i] / (ssum[(size_t)d*4 + i] + 1e-16f);
        atomicAdd(&agg[(size_t)d * HID_C + c], hsrc[(size_t)s * HID_C + c] * wgt);
    }
}

__global__ void relu_ip(float* __restrict__ p, size_t n) {
    size_t i = (size_t)blockIdx.x * blockDim.x + threadIdx.x;
    if (i < n) p[i] = fmaxf(p[i], 0.0f);
}
__global__ void fill_f32(float* __restrict__ p, float v, size_t n) {
    size_t i = (size_t)blockIdx.x * blockDim.x + threadIdx.x;
    if (i < n) p[i] = v;
}
__global__ void fill_u32(unsigned* __restrict__ p, unsigned v, size_t n) {
    size_t i = (size_t)blockIdx.x * blockDim.x + threadIdx.x;
    if (i < n) p[i] = v;
}

// semantic attention combine over 2 metapaths: out = w0*a + w1*b, w = softmax(score/Nn)
__global__ void combine2(const float* __restrict__ a, const float* __restrict__ b,
                         const float* __restrict__ sc, float invN,
                         float* __restrict__ out, size_t n)
{
    size_t i = (size_t)blockIdx.x * blockDim.x + threadIdx.x;
    if (i >= n) return;
    float s0 = sc[0] * invN, s1 = sc[1] * invN;
    float m  = fmaxf(s0, s1);
    float e0 = __expf(s0 - m), e1 = __expf(s1 - m);
    float inv = 1.0f / (e0 + e1);
    out[i] = (e0 * inv) * a[i] + (e1 * inv) * b[i];
}

// context: scan co edges whose src is in masked set; accumulate dst embeddings
__global__ void context_scan(const int* __restrict__ eco, int E,
                             const int* __restrict__ masked,
                             const float* __restrict__ x2i, float* __restrict__ acc)
{
    __shared__ int mk[K_MASK];
    if (threadIdx.x < K_MASK) mk[threadIdx.x] = masked[threadIdx.x];
    __syncthreads();
    int e = blockIdx.x * blockDim.x + threadIdx.x;
    if (e >= E) return;
    int s = eco[e];
    bool hit = false;
    #pragma unroll
    for (int i = 0; i < K_MASK; ++i) hit = hit || (s == mk[i]);
    if (hit) {
        atomicAdd(&acc[HID_C], 1.0f);
        int d = eco[E + e];
        const float* xp = x2i + (size_t)d * HID_C;
        for (int c = 0; c < HID_C; ++c) atomicAdd(&acc[c], xp[c]);
    }
}
__global__ void context_fin(const float* __restrict__ acc, float* __restrict__ out) {
    int c = threadIdx.x;
    float den = acc[HID_C];
    out[c] = den > 0.0f ? acc[c] / fmaxf(den, 1.0f) : 0.0f;
}

// ---------------- host-side GEMM dispatch ----------------
static inline void launch_gemm(int act, const float* A, const float* B, const float* bias,
                               float* C, int M, int N, int Kd, const int* rowidx,
                               const float* qv, float* score, hipStream_t s)
{
    dim3 g((unsigned)((N + 63) / 64), (unsigned)((M + 63) / 64)), b(256);
    if (act == 0)      gemm_wmma_bf16<0><<<g, b, 0, s>>>(A, B, bias, C, M, N, Kd, rowidx, qv, score);
    else if (act == 1) gemm_wmma_bf16<1><<<g, b, 0, s>>>(A, B, bias, C, M, N, Kd, rowidx, qv, score);
    else               gemm_wmma_bf16<2><<<g, b, 0, s>>>(A, B, bias, C, M, N, Kd, rowidx, qv, score);
}

extern "C" void kernel_launch(void* const* d_in, const int* in_sizes, int n_in,
                              void* d_out, int out_size, void* d_ws, size_t ws_size,
                              hipStream_t stream)
{
    // Input flattening follows jax pytree order (dict keys sorted):
    // 0 x_ingre, 1 x_direc, 2 ei_co, 3 ei_ui, 4 ei_pw, 5 masked_idx,
    // params: fc_logvar{b,w}, fc_mu{b,w},
    //   han1: att{co{dst,src},pw{dst,src},ui{dst,src}}, k_lin{b,w},
    //         proj{direction{b,w},ingredient{b,w}}, q            (indices 10..22)
    //   han2: same layout                                         (indices 23..35)
    const float* x_ingre = (const float*)d_in[0];
    const float* x_direc = (const float*)d_in[1];
    const int*   ei_co   = (const int*)d_in[2];
    const int*   ei_ui   = (const int*)d_in[3];
    const int*   ei_pw   = (const int*)d_in[4];
    const int*   masked  = (const int*)d_in[5];
    const float* fc_lv_b = (const float*)d_in[6];
    const float* fc_lv_w = (const float*)d_in[7];
    const float* fc_mu_b = (const float*)d_in[8];
    const float* fc_mu_w = (const float*)d_in[9];

    struct Han {
        const float *co_dst, *co_src, *pw_dst, *pw_src, *ui_dst, *ui_src;
        const float *kb, *kw, *pdb, *pdw, *pib, *piw, *q;
    };
    auto loadHan = [&](int base) {
        Han h;
        h.co_dst = (const float*)d_in[base+0];  h.co_src = (const float*)d_in[base+1];
        h.pw_dst = (const float*)d_in[base+2];  h.pw_src = (const float*)d_in[base+3];
        h.ui_dst = (const float*)d_in[base+4];  h.ui_src = (const float*)d_in[base+5];
        h.kb  = (const float*)d_in[base+6];     h.kw  = (const float*)d_in[base+7];
        h.pdb = (const float*)d_in[base+8];     h.pdw = (const float*)d_in[base+9];
        h.pib = (const float*)d_in[base+10];    h.piw = (const float*)d_in[base+11];
        h.q   = (const float*)d_in[base+12];
        return h;
    };
    Han h1 = loadHan(10), h2 = loadHan(23);

    // ---------------- workspace carving (floats) ----------------
    float* ws = (float*)d_ws;
    size_t off = 0;
    auto alloc = [&](size_t n) { float* p = ws + off; off += (n + 127) & ~(size_t)127; return p; };
    float*    h_i    = alloc((size_t)NI_N * HID_C);
    float*    h_d    = alloc((size_t)ND_N * HID_C);
    float*    agg_co = alloc((size_t)NI_N * HID_C);
    float*    agg_pw = alloc((size_t)NI_N * HID_C);
    float*    agg_ui = alloc((size_t)ND_N * HID_C);
    float*    x1_i   = alloc((size_t)NI_N * HID_C);
    float*    x1_d   = alloc((size_t)ND_N * HID_C);
    float*    asrc   = alloc((size_t)NI_N * 4);
    float*    adst   = alloc((size_t)NI_N * 4);
    unsigned* amax   = (unsigned*)alloc((size_t)NI_N * 4);
    float*    ssum   = alloc((size_t)NI_N * 4);
    float*    ebuf   = alloc((size_t)E_CO_N * 4);
    float*    scores = alloc(128);
    float*    ctx    = alloc(256);
    float*    x2_i   = x1_i;   // safe reuse: x1 dead once layer-2 projections done

    auto fillf = [&](float* p, float v, size_t n) {
        fill_f32<<<(unsigned)((n + 255) / 256), 256, 0, stream>>>(p, v, n);
    };
    auto fillu = [&](unsigned* p, unsigned v, size_t n) {
        fill_u32<<<(unsigned)((n + 255) / 256), 256, 0, stream>>>(p, v, n);
    };

    auto edge_pass = [&](const int* ei, int E, const float* hs, int Ns,
                         const float* hdd, int Nd,
                         const float* attS, const float* attD, float* agg) {
        head_score<<<(Ns * 4 + 255) / 256, 256, 0, stream>>>(hs,  attS, asrc, Ns);
        head_score<<<(Nd * 4 + 255) / 256, 256, 0, stream>>>(hdd, attD, adst, Nd);
        fillu(amax, ENC_NEG_INF, (size_t)Nd * 4);
        fillf(ssum, 0.0f,        (size_t)Nd * 4);
        edge_alpha_max<<<(E + 255) / 256, 256, 0, stream>>>(ei, E, asrc, adst, ebuf, amax);
        edge_exp_sum  <<<(E + 255) / 256, 256, 0, stream>>>(ei, E, ebuf, amax, ssum);
        fillf(agg, 0.0f, (size_t)Nd * HID_C);
        edge_scatter<<<(unsigned)((E + 7) / 8), 256, 0, stream>>>(ei, E, hs, ebuf, ssum, agg);
        relu_ip<<<(unsigned)(((size_t)Nd * HID_C + 255) / 256), 256, 0, stream>>>(agg, (size_t)Nd * HID_C);
    };

    auto layer = [&](const float* xi, int Ki, const float* xdl, int Kdm, const Han& P,
                     bool dirOut, float* out_i, float* out_d) {
        // node-type projections (WMMA GEMMs)
        launch_gemm(0, xi,  P.piw, P.pib, h_i, NI_N, HID_C, Ki,  nullptr, nullptr, nullptr, stream);
        launch_gemm(0, xdl, P.pdw, P.pdb, h_d, ND_N, HID_C, Kdm, nullptr, nullptr, nullptr, stream);
        // metapath message passing
        edge_pass(ei_co, E_CO_N, h_i, NI_N, h_i, NI_N, P.co_src, P.co_dst, agg_co);
        if (dirOut)
            edge_pass(ei_ui, E_UI_N, h_i, NI_N, h_d, ND_N, P.ui_src, P.ui_dst, agg_ui);
        edge_pass(ei_pw, E_PW_N, h_d, ND_N, h_i, NI_N, P.pw_src, P.pw_dst, agg_pw);
        // semantic attention scores (fused tanh(o@kW+kb)·q reduction, WMMA)
        fillf(scores, 0.0f, 4);
        launch_gemm(2, agg_co, P.kw, P.kb, nullptr, NI_N, HID_C, HID_C, nullptr, P.q, scores + 0, stream);
        launch_gemm(2, agg_pw, P.kw, P.kb, nullptr, NI_N, HID_C, HID_C, nullptr, P.q, scores + 1, stream);
        combine2<<<(unsigned)(((size_t)NI_N * HID_C + 255) / 256), 256, 0, stream>>>(
            agg_co, agg_pw, scores, 1.0f / (float)NI_N, out_i, (size_t)NI_N * HID_C);
        if (dirOut)   // single metapath -> softmax weight == 1 -> plain copy
            hipMemcpyAsync(out_d, agg_ui, (size_t)ND_N * HID_C * sizeof(float),
                           hipMemcpyDeviceToDevice, stream);
    };

    // layer 1: in dims 128 / 768; layer 2: 128 / 128 (direction output dead)
    layer(x_ingre, 128, x_direc, 768, h1, true,  x1_i, x1_d);
    layer(x1_i,    128, x1_d,    128, h2, false, x2_i, nullptr);

    // ---------------- heads: mu / logvar on gathered masked rows ----------------
    float* mu  = (float*)d_out;
    float* lv  = mu + K_MASK * LAT_N;
    float* cto = mu + 2 * K_MASK * LAT_N;
    launch_gemm(0, x2_i, fc_mu_w, fc_mu_b, mu, K_MASK, LAT_N, HID_C, masked, nullptr, nullptr, stream);
    launch_gemm(0, x2_i, fc_lv_w, fc_lv_b, lv, K_MASK, LAT_N, HID_C, masked, nullptr, nullptr, stream);

    // ---------------- neighbor context over co edges ----------------
    fillf(ctx, 0.0f, 132);
    context_scan<<<(E_CO_N + 255) / 256, 256, 0, stream>>>(ei_co, E_CO_N, masked, x2_i, ctx);
    context_fin<<<1, 128, 0, stream>>>(ctx, cto);
}
